// DiagonalBiLSTM_76587856822423
// MI455X (gfx1250) — matrix operand order repaired and checked by reading, not compiled, not run on hardware
//
#include <hip/hip_runtime.h>
#include <hip/hip_bf16.h>

typedef __attribute__((ext_vector_type(16))) _Float16 v16h;
typedef __attribute__((ext_vector_type(8)))  _Float16 v8h;
typedef __attribute__((ext_vector_type(8)))  float    v8f;

#define DEV __device__ __forceinline__

DEV float sigf(float x) { return 1.0f / (1.0f + __expf(-x)); }

// ---- CDNA5 wave32 WMMA fragment helpers (v_wmma_f32_16x16x32_f16) ----
// A 16x32 (MxK) f16 from row-major source (ld elems): two contiguous 16B chunks per lane.
DEV v16h load_a_frag(const _Float16* p, int ld, int lane) {
  int r  = lane & 15;
  int lo = (lane & 16) ? 8 : 0;
  const _Float16* q = p + (size_t)r * ld + lo;
  v8h l8 = *(const v8h*)q;
  v8h h8 = *(const v8h*)(q + 16);
  v16h a;
#pragma unroll
  for (int i = 0; i < 8; ++i) { a[i] = l8[i]; a[i + 8] = h8[i]; }
  return a;
}
// B 32x16 fragments pre-swizzled: frag f -> 32 lanes x 16 contiguous halves.
DEV v16h load_b_sw(const _Float16* Bsw, int frag, int lane) {
  return *(const v16h*)(Bsw + ((size_t)frag * 32 + lane) * 16);
}
// C/D f32 16x16: lane col = lane&15; vgpr g holds row g + ((lane&16)?8:0)

// Swizzle f32 row-major B (K x N) into f16 per-lane fragment layout.
__global__ __launch_bounds__(256) void swizzle_b_kernel(const float* __restrict__ B,
                                                        _Float16* __restrict__ Bsw,
                                                        int K, int N) {
  int idx = blockIdx.x * 256 + threadIdx.x;
  int total = (K >> 5) * (N >> 4) * 512;
  if (idx >= total) return;
  int j    = idx & 15;
  int lane = (idx >> 4) & 31;
  int tile = idx >> 9;
  int tiles_n = N >> 4;
  int kt = tile / tiles_n, ct = tile % tiles_n;
  int k   = (kt << 5) + ((lane & 16) ? 8 : 0) + (j & 7) + ((j & 8) ? 16 : 0);
  int col = (ct << 4) + (lane & 15);
  Bsw[idx] = (_Float16)B[(size_t)k * N + col];
}

// ---------------- generic GEMM (head): C = A(rows x K f16) * Bsw + bias ----------------
template <int K>
__global__ __launch_bounds__(256) void gemm_f16_kernel(
    const _Float16* __restrict__ A, const _Float16* __restrict__ Bsw,
    const float* __restrict__ bias0,
    float* __restrict__ Cf, _Float16* __restrict__ Ch, int rows, int N) {
  int lane = threadIdx.x & 31;
  int wave = threadIdx.x >> 5;
  int tiles_n = N >> 4;
  int total = (rows >> 4) * tiles_n;
  int tile = blockIdx.x * 8 + wave;
  if (tile >= total) return;  // wave-uniform
  int rt = tile / tiles_n, ct = tile % tiles_n;
  int row0 = rt << 4, col0 = ct << 4;
  v8f acc = {};
#pragma unroll
  for (int kt = 0; kt < K / 32; ++kt) {
    v16h a  = load_a_frag(A + (size_t)row0 * K + kt * 32, K, lane);
    v16h bf = load_b_sw(Bsw, kt * tiles_n + ct, lane);
    acc = __builtin_amdgcn_wmma_f32_16x16x32_f16(false, a, false, bf, (short)0, acc, false, false);
  }
  int col = col0 + (lane & 15);
  float bv = bias0 ? bias0[col] : 0.0f;
  int roff = (lane & 16) ? 8 : 0;
#pragma unroll
  for (int g = 0; g < 8; ++g) {
    float v = acc[g] + bv;
    size_t idx = (size_t)(row0 + roff + g) * N + col;
    if (Cf) Cf[idx] = v;
    if (Ch) Ch[idx] = (_Float16)v;
  }
}

// ---------------- zx projection, fragment-layout output ----------------
// zx_sw[((b*127+t)*64 + rt*16+ct)*32*8 ...] = C-fragments of xn @ w_is + b_is + b_ss
__global__ __launch_bounds__(256) void zx_gemm_kernel(
    const _Float16* __restrict__ xn16,   // (B,64,127,128) row-major
    const _Float16* __restrict__ Bsw,    // w_is swizzled 128x256
    const float* __restrict__ bias0, const float* __restrict__ bias1,
    float* __restrict__ zx_sw) {
  int lane = threadIdx.x & 31, wave = threadIdx.x >> 5;
  int tg = blockIdx.x * 8 + wave;        // bt*64 + rt*16 + ct
  int ct = tg & 15;
  int rt = (tg >> 4) & 3;
  int bt = tg >> 6;                      // b*127 + t
  int b = bt / 127, t = bt % 127;
  const _Float16* Abase = xn16 + ((size_t)(b * 64 + rt * 16) * 127 + t) * 128;
  v8f acc = {};
#pragma unroll
  for (int kt = 0; kt < 4; ++kt) {
    v16h a  = load_a_frag(Abase + kt * 32, 127 * 128, lane);
    v16h bf = load_b_sw(Bsw, kt * 16 + ct, lane);
    acc = __builtin_amdgcn_wmma_f32_16x16x32_f16(false, a, false, bf, (short)0, acc, false, false);
  }
  int col = (ct << 4) + (lane & 15);
  float bv = bias0[col] + bias1[col];
#pragma unroll
  for (int g = 0; g < 8; ++g) acc[g] += bv;
  *(v8f*)(zx_sw + ((size_t)tg * 32 + lane) * 8) = acc;
}

// ---------------- masked 7x7 conv on skewed image ----------------
__global__ __launch_bounds__(256) void conv_skew_kernel(
    const float* __restrict__ im, const float* __restrict__ wk,
    const float* __restrict__ wb, float* __restrict__ xout) {
  int idx = blockIdx.x * 256 + threadIdx.x;
  int co = idx & 127;
  int t  = idx >> 7;
  int j  = t % 127; t /= 127;
  int i  = t & 63;
  int b  = t >> 6;
  float acc = wb[co];
  for (int kh = 0; kh < 4; ++kh) {
    int r = i + kh - 3;
    if (r < 0 || r >= 64) continue;
    int kwmax = (kh == 3) ? 3 : 7;
    for (int kw = 0; kw < kwmax; ++kw) {
      int cc = j + kw - 3;
      int n = cc - r;
      if (cc < 0 || cc >= 127 || n < 0 || n >= 64) continue;
      acc += im[(size_t)(b * 64 + r) * 64 + n] * wk[(size_t)(kh * 7 + kw) * 128 + co];
    }
  }
  xout[idx] = acc;
}

// ---------------- LayerNorm (128 ch), one wave per row; row-major + residual fragments ----
__global__ __launch_bounds__(256) void ln_kernel(
    const float* __restrict__ x, const float* __restrict__ s,
    const float* __restrict__ bta, _Float16* __restrict__ xn,
    _Float16* __restrict__ xnr_sw) {
  int lane = threadIdx.x & 31;
  int row  = blockIdx.x * 8 + (threadIdx.x >> 5);
  const float* p = x + (size_t)row * 128;
  float v[4];
  float sum = 0.0f, sq = 0.0f;
#pragma unroll
  for (int i = 0; i < 4; ++i) { v[i] = p[lane * 4 + i]; sum += v[i]; sq += v[i] * v[i]; }
#pragma unroll
  for (int o = 16; o > 0; o >>= 1) { sum += __shfl_xor(sum, o, 32); sq += __shfl_xor(sq, o, 32); }
  float mu  = sum * (1.0f / 128.0f);
  float var = sq * (1.0f / 128.0f) - mu * mu;
  float inv = rsqrtf(var + 1e-6f);
  _Float16* q = xn + (size_t)row * 128;
  // row -> (b,m,t) ; residual fragment coords
  int r2 = row;
  int t  = r2 % 127; r2 /= 127;
  int m  = r2 & 63;
  int bb = r2 >> 6;
  size_t bt = (size_t)bb * 127 + t;
  int lh = (m >> 3) & 1, g = m & 7, rt = m >> 4;
  _Float16* base = xnr_sw + bt * 8192;
#pragma unroll
  for (int i = 0; i < 4; ++i) {
    int ch = lane * 4 + i;
    _Float16 hv = (_Float16)((v[i] - mu) * inv * s[ch] + bta[ch]);
    q[ch] = hv;
    base[((rt * 8 + (ch >> 4)) * 32 + lh * 16 + (ch & 15)) * 8 + g] = hv;
  }
}

__global__ __launch_bounds__(256) void zero_kernel(float* __restrict__ p, int n) {
  int i = blockIdx.x * 256 + threadIdx.x;
  if (i < n) p[i] = 0.0f;
}

// xu[b,m,j,k] = x[b,m,j+m,k]  (unskew) -> f16
__global__ __launch_bounds__(256) void unskew_f16_kernel(const float* __restrict__ x,
                                                         _Float16* __restrict__ xu) {
  int idx = blockIdx.x * 256 + threadIdx.x;
  int k = idx & 127;
  int t = idx >> 7;
  int j = t & 63; t >>= 6;
  int m = t & 63;
  int b = t >> 6;
  xu[idx] = (_Float16)x[((size_t)(b * 64 + m) * 127 + (j + m)) * 128 + k];
}

// ---------------- persistent diagonal-LSTM scan: one workgroup per batch element ------------
// Wave w owns: row-tile pair rbase=2*(w>>2), column-tile group {c, c+4, c+8, c+12} with c=w&3,
// so all four LSTM gate quarters for a feature land in the same wave/lane -> gates in registers.
// A2_s has a permanent zero row 0 and A2_s[m+1] = h[m]:  hprev frag = rows row0, h frag = rows row0+1.
__global__ __launch_bounds__(256) void scan_kernel(
    const float* __restrict__ zx_sw,      // (B*127, 64 tiles, 32, 8) f32 C-fragments
    const _Float16* __restrict__ xnr_sw,  // (B*127, 32 tiles, 32, 8) f16 residual fragments
    const _Float16* __restrict__ wcat_sw, // 128x256 swizzled [wss0|wss1]
    const _Float16* __restrict__ woc_sw,  // 64x128 swizzled
    const float* __restrict__ boc, const float* __restrict__ h0p,
    float* __restrict__ xnext, int dir) {
  __shared__ __align__(32) _Float16 A2_s[65 * 64];  // 8.3 KB total LDS

  int tid = threadIdx.x, lane = tid & 31, wave = tid >> 5;
  int b = blockIdx.x;
  int roff  = (lane & 16) ? 8 : 0;
  int rbase = (wave >> 2) * 2;
  int cgrp  = wave & 3;

  // resident B fragments
  v16h zb[4][4];
#pragma unroll
  for (int kt = 0; kt < 4; ++kt)
#pragma unroll
    for (int cg = 0; cg < 4; ++cg)
      zb[kt][cg] = load_b_sw(wcat_sw, kt * 16 + cgrp + cg * 4, lane);
  v16h ob0 = load_b_sw(woc_sw, wave, lane);
  v16h ob1 = load_b_sw(woc_sw, 8 + wave, lane);
  int  col_oc = (wave << 4) + (lane & 15);
  float bv_oc = boc[col_oc];

  for (int e = tid; e < 65 * 64; e += 256) {
    int r = e >> 6, f = e & 63;
    A2_s[e] = (r == 0) ? (_Float16)0.0f : (_Float16)h0p[f];  // h = h0 broadcast; row0 = 0 forever
  }
  float c_reg[2][8];
#pragma unroll
  for (int rr = 0; rr < 2; ++rr)
#pragma unroll
    for (int g = 0; g < 8; ++g) c_reg[rr][g] = 0.0f;
  __syncthreads();

  for (int s = 0; s < 127; ++s) {
    int t = dir ? (126 - s) : s;
    size_t bt = (size_t)b * 127 + t;
    const float*    zxt = zx_sw + bt * 16384;   // 64 tiles * 256
    const _Float16* xnt = xnr_sw + bt * 8192;   // 32 tiles * 256

    // z-GEMM: z = zx[t] + [hprev|h] @ wcat, accumulators stay in registers
    v8f zacc[2][4];
#pragma unroll
    for (int rr = 0; rr < 2; ++rr) {
      int row0 = (rbase + rr) << 4;
      v16h a0 = load_a_frag(A2_s + (row0)     * 64 +  0, 64, lane);  // hprev K 0..31
      v16h a1 = load_a_frag(A2_s + (row0)     * 64 + 32, 64, lane);  // hprev K 32..63
      v16h a2 = load_a_frag(A2_s + (row0 + 1) * 64 +  0, 64, lane);  // h     K 64..95
      v16h a3 = load_a_frag(A2_s + (row0 + 1) * 64 + 32, 64, lane);  // h     K 96..127
#pragma unroll
      for (int cg = 0; cg < 4; ++cg) {
        int ct = cgrp + cg * 4;
        v8f acc = *(const v8f*)(zxt + (((rbase + rr) * 16 + ct) * 32 + lane) * 8);
        acc = __builtin_amdgcn_wmma_f32_16x16x32_f16(false, a0, false, zb[0][cg], (short)0, acc, false, false);
        acc = __builtin_amdgcn_wmma_f32_16x16x32_f16(false, a1, false, zb[1][cg], (short)0, acc, false, false);
        acc = __builtin_amdgcn_wmma_f32_16x16x32_f16(false, a2, false, zb[2][cg], (short)0, acc, false, false);
        acc = __builtin_amdgcn_wmma_f32_16x16x32_f16(false, a3, false, zb[3][cg], (short)0, acc, false, false);
        zacc[rr][cg] = acc;
      }
    }
    __syncthreads();  // all A2 reads complete before h update

    // gates fully in registers: cg0=forget, cg1=input, cg2=output, cg3=g
    int fcol = (cgrp << 4) + (lane & 15);
#pragma unroll
    for (int rr = 0; rr < 2; ++rr) {
      int row0 = (rbase + rr) << 4;
#pragma unroll
      for (int g = 0; g < 8; ++g) {
        int m = row0 + roff + g;
        float cn = sigf(zacc[rr][0][g]) * c_reg[rr][g]
                 + sigf(zacc[rr][1][g]) * tanhf(zacc[rr][3][g]);
        c_reg[rr][g] = cn;
        A2_s[(m + 1) * 64 + fcol] = (_Float16)(sigf(zacc[rr][2][g]) * tanhf(cn));
      }
    }
    __syncthreads();  // new h visible

    // x_out = hn @ woc + boc + xn ; accumulate (2 contributions -> deterministic)
#pragma unroll
    for (int rt = 0; rt < 4; ++rt) {
      int row0 = rt << 4;
      v8h xr = *(const v8h*)(xnt + ((rt * 8 + wave) * 32 + lane) * 8);
      v8f acc;
#pragma unroll
      for (int g = 0; g < 8; ++g) acc[g] = bv_oc + (float)xr[g];
      v16h a0 = load_a_frag(A2_s + (row0 + 1) * 64 +  0, 64, lane);
      v16h a1 = load_a_frag(A2_s + (row0 + 1) * 64 + 32, 64, lane);
      acc = __builtin_amdgcn_wmma_f32_16x16x32_f16(false, a0, false, ob0, (short)0, acc, false, false);
      acc = __builtin_amdgcn_wmma_f32_16x16x32_f16(false, a1, false, ob1, (short)0, acc, false, false);
#pragma unroll
      for (int g = 0; g < 8; ++g) {
        int m = row0 + roff + g;
        int mo = m + dir;
        if (mo < 64)
          atomicAdd(&xnext[((size_t)(b * 64 + mo) * 127 + t) * 128 + col_oc], acc[g]);
      }
    }
    // no barrier needed: next z-GEMM only reads A2; writes happen after next barrier
  }
}

// ------------------------------- host orchestration -------------------------------
extern "C" void kernel_launch(void* const* d_in, const int* in_sizes, int n_in,
                              void* d_out, int out_size, void* d_ws, size_t ws_size,
                              hipStream_t stream) {
  const float* im     = (const float*)d_in[0];
  const float* conv_k = (const float*)d_in[1];
  const float* conv_b = (const float*)d_in[2];
  const float* ln_s   = (const float*)d_in[3];
  const float* ln_b   = (const float*)d_in[4];
  const float* w_is   = (const float*)d_in[5];
  const float* b_is   = (const float*)d_in[6];
  const float* w_ss   = (const float*)d_in[7];
  const float* b_ss   = (const float*)d_in[8];
  const float* w_oc   = (const float*)d_in[9];
  const float* b_oc   = (const float*)d_in[10];
  const float* h0     = (const float*)d_in[11];
  const float* w_out1 = (const float*)d_in[12];
  const float* b_out1 = (const float*)d_in[13];
  const float* w_out2 = (const float*)d_in[14];
  const float* b_out2 = (const float*)d_in[15];
  const float* w_head = (const float*)d_in[16];
  const float* b_head = (const float*)d_in[17];

  const int B = 16, M = 64, N2 = 127, H2 = 128, F4 = 256;
  const size_t nX  = (size_t)B * M * N2 * H2;
  const size_t nZX = (size_t)B * M * N2 * F4;
  const size_t nU  = (size_t)B * M * 64 * H2;

  char* w = (char*)d_ws;
  auto carve = [&](size_t bytes) -> void* {
    void* p = (void*)w;
    w += (bytes + 255) & ~(size_t)255;
    return p;
  };
  float*    xA      = (float*)carve(nX * 4);
  float*    xB      = (float*)carve(nX * 4);
  _Float16* xn16    = (_Float16*)carve(nX * 2);
  _Float16* xnr_sw  = (_Float16*)carve(nX * 2);
  float*    zx_sw   = (float*)carve(nZX * 4);
  _Float16* xu16    = (_Float16*)carve(nU * 2);
  _Float16* y1_16   = (_Float16*)carve((size_t)65536 * 32 * 2);
  _Float16* y2_16   = (_Float16*)carve((size_t)65536 * 32 * 2);
  _Float16* wis_sw  = (_Float16*)carve(32768 * 2);
  _Float16* wcat_sw = (_Float16*)carve(32768 * 2);
  _Float16* woc_sw  = (_Float16*)carve(8192 * 2);
  _Float16* wo1_sw  = (_Float16*)carve(4096 * 2);
  _Float16* wo2_sw  = (_Float16*)carve(1024 * 2);
  _Float16* wh_sw   = (_Float16*)carve(8192 * 2);

  auto swz = [&](const float* src, _Float16* dst, int K, int N) {
    int total = (K / 32) * (N / 16) * 512;
    swizzle_b_kernel<<<(total + 255) / 256, 256, 0, stream>>>(src, dst, K, N);
  };

  swz(w_out1, wo1_sw, 128, 32);
  swz(w_out2, wo2_sw, 32, 32);
  swz(w_head, wh_sw, 32, 256);

  conv_skew_kernel<<<(int)(nX / 256), 256, 0, stream>>>(im, conv_k, conv_b, xA);

  const int rowsX = B * M * N2;  // 130048
  for (int l = 0; l < 2; ++l) {
    float* xin  = (l == 0) ? xA : xB;
    float* xout = (l == 0) ? xB : xA;
    zero_kernel<<<(int)(nX / 256), 256, 0, stream>>>(xout, (int)nX);
    for (int dir = 0; dir < 2; ++dir) {
      int ld = l * 2 + dir;
      ln_kernel<<<rowsX / 8, 256, 0, stream>>>(xin, ln_s + ld * 128, ln_b + ld * 128,
                                               xn16, xnr_sw);
      swz(w_is + (size_t)ld * 32768, wis_sw, 128, 256);
      swz(w_ss + (size_t)ld * 32768, wcat_sw, 128, 256);
      swz(w_oc + (size_t)ld * 8192, woc_sw, 64, 128);
      // hoisted input projection in fragment layout
      zx_gemm_kernel<<<(B * N2 * 64) / 8, 256, 0, stream>>>(
          xn16, wis_sw, b_is + ld * 256, b_ss + ld * 256, zx_sw);
      // persistent recurrent scan
      scan_kernel<<<B, 256, 0, stream>>>(zx_sw, xnr_sw, wcat_sw, woc_sw,
                                         b_oc + ld * 128, h0 + ld * 64, xout, dir);
    }
  }

  // unskew + output head: 128 -> 32 -> 32 -> 256
  unskew_f16_kernel<<<(int)(nU / 256), 256, 0, stream>>>(xA, xu16);
  {
    int rows = 65536;
    gemm_f16_kernel<128><<<(rows / 16) * 2 / 8, 256, 0, stream>>>(
        xu16, wo1_sw, b_out1, nullptr, y1_16, rows, 32);
    gemm_f16_kernel<32><<<(rows / 16) * 2 / 8, 256, 0, stream>>>(
        y1_16, wo2_sw, b_out2, nullptr, y2_16, rows, 32);
    gemm_f16_kernel<32><<<(rows / 16) * 16 / 8, 256, 0, stream>>>(
        y2_16, wh_sw, b_head, (float*)d_out, nullptr, rows, 256);
  }
}
